// ConvModule_9165460209876
// MI455X (gfx1250) — matrix operand order repaired and checked by reading, not compiled, not run on hardware
//
#include <hip/hip_runtime.h>
#include <hip/hip_bf16.h>

#define BSZ 32
#define NN  512
#define LL  4096
#define HH  256
#define EE  64
#define WW  8
#define KK  9
#define NEE 2048
#define CIN 392
#define CINP 416      // 13*32
#define KT1 13
#define KT2 8
#define CSH 4
#define LCC 4088
#define NSEG 511
#define EPSF 1e-5f

typedef __attribute__((ext_vector_type(16))) _Float16 v16h;
typedef __attribute__((ext_vector_type(8)))  _Float16 v8h;
typedef __attribute__((ext_vector_type(8)))  float    v8f;

// ---- WMMA fragment loads (contiguous k order, same for A and B) ----
// A 16x32 f16: lane<16 holds row M=lane, K=0..15; lane>=16 holds row lane-16, K=16..31
__device__ __forceinline__ v16h load_frag_a(const _Float16* base, int half) {
    return *(const v16h*)(base + half * 16);
}
// B 32x16 f16: lane<16 holds col N=lane, K=0..15; lane>=16 holds col lane-16, K=16..31
__device__ __forceinline__ v16h load_frag_b(const _Float16* base, int half) {
    return *(const v16h*)(base + half * 16);
}
__device__ __forceinline__ v8f wmma_f16(v16h a, v16h b, v8f c) {
    return __builtin_amdgcn_wmma_f32_16x16x32_f16(false, a, false, b, (short)0, c, false, false);
}

// ---- 1. LayerNorm over H ----
__global__ void k_layernorm(const float* __restrict__ x, const float* __restrict__ g,
                            const float* __restrict__ bt, float* __restrict__ xn) {
    int row = blockIdx.x;          // b*N + n
    int t = threadIdx.x;           // 256 == H
    float v = x[(size_t)row * HH + t];
    __shared__ float s1[HH], s2[HH];
    s1[t] = v; s2[t] = v * v; __syncthreads();
    for (int s = HH / 2; s > 0; s >>= 1) {
        if (t < s) { s1[t] += s1[t + s]; s2[t] += s2[t + s]; }
        __syncthreads();
    }
    float mu = s1[0] / HH;
    float var = s2[0] / HH - mu * mu;
    float r = rsqrtf(var + EPSF);
    xn[(size_t)row * HH + t] = (v - mu) * r * g[t] + bt[t];
}

// ---- 2. Pack pc1_w (512 x 392 -> 32x13x16x32 f16, zero-padded) ----
__global__ void k_pack_w1(const float* __restrict__ w, _Float16* __restrict__ Ap) {
    int i = blockIdx.x * blockDim.x + threadIdx.x;
    if (i >= 32 * KT1 * 16 * 32) return;
    int e = i & 31, r = (i >> 5) & 15, kt = (i / 512) % KT1, mt = i / (512 * KT1);
    int row = mt * 16 + r, col = kt * 32 + e;
    float v = (col < CIN) ? w[(size_t)row * CIN + col] : 0.f;
    Ap[i] = (_Float16)v;
}

// ---- 3. Pack pc2_w (256 x 256 -> 16x8x16x32 f16) ----
__global__ void k_pack_w2(const float* __restrict__ w, _Float16* __restrict__ Ap) {
    int i = blockIdx.x * blockDim.x + threadIdx.x;
    if (i >= 16 * KT2 * 16 * 32) return;
    int e = i & 31, r = (i >> 5) & 15, kt = (i / 512) % KT2, mt = i / (512 * KT2);
    Ap[i] = (_Float16)w[(size_t)(mt * 16 + r) * HH + kt * 32 + e];
}

// ---- 4. Gather + concat -> packed f16 input [b][kt][l][32] ----
__global__ void k_build_h(const float* __restrict__ xn, const float* __restrict__ ef,
                          const int* __restrict__ wn, const int* __restrict__ we,
                          const float* __restrict__ enc, _Float16* __restrict__ Hp) {
    int bl = blockIdx.x;
    int b = bl / LL, l = bl % LL;
    int node = wn[b * LL + l] + 1;
    const float* xrow = xn + ((size_t)b * NN + node) * HH;
    for (int c = threadIdx.x; c < CINP; c += blockDim.x) {
        float v;
        if (c < HH) v = xrow[c];
        else if (c < HH + EE)
            v = (l == 0) ? 0.f : ef[((size_t)b * NEE + we[b * (LL - 1) + (l - 1)]) * EE + (c - HH)];
        else if (c < HH + 2 * EE)
            v = (l == LL - 1) ? 0.f : ef[((size_t)b * NEE + we[b * (LL - 1) + l]) * EE + (c - HH - EE)];
        else if (c < CIN)
            v = enc[((size_t)b * WW + (c - HH - 2 * EE)) * LL + l];
        else v = 0.f;
        Hp[(((size_t)b * KT1 + (c >> 5)) * LL + l) * 32 + (c & 31)] = (_Float16)v;
    }
}

// ---- 5. GEMM1 (WMMA f16) fused with bias + GLU ----
// block: 128 threads = 4 waves; 4 waves share one N-tile (B-frag reuse in L1)
__global__ void k_gemm1_glu(const _Float16* __restrict__ Ap, const _Float16* __restrict__ Hp,
                            const float* __restrict__ b1, float* __restrict__ h1) {
    int lane = threadIdx.x & 31, wv = threadIdx.x >> 5;
    int bid = blockIdx.x;
    int nt = bid & 255, mtg = (bid >> 8) & 3, b = bid >> 10;
    int mt = mtg * 4 + wv;             // 0..15 (out-row tile; gate tile = mt+16)
    int half = lane >> 4, lr = lane & 15;
    int col = nt * 16 + lr;
    v8f accO = {}, accG = {};
    const _Float16* Bb = Hp + (((size_t)b * KT1) * LL + col) * 32;
    const _Float16* Ao = Ap + (((size_t)mt * KT1) * 16 + lr) * 32;
    const _Float16* Ag = Ap + (((size_t)(mt + 16) * KT1) * 16 + lr) * 32;
    for (int kt = 0; kt < KT1; ++kt) {
        if (kt + 1 < KT1) __builtin_prefetch(Bb + (size_t)(kt + 1) * LL * 32, 0, 1);
        v16h bf = load_frag_b(Bb + (size_t)kt * LL * 32, half);
        v16h ao = load_frag_a(Ao + kt * 512, half);
        v16h ag = load_frag_a(Ag + kt * 512, half);
        accO = wmma_f16(ao, bf, accO);
        accG = wmma_f16(ag, bf, accG);
    }
#pragma unroll
    for (int j = 0; j < 8; ++j) {
        int row = mt * 16 + j + 8 * half;
        float o = accO[j] + b1[row];
        float gt = accG[j] + b1[row + 256];
        float s = 1.f / (1.f + __expf(-gt));
        h1[((size_t)b * HH + row) * LL + col] = o * s;
    }
}

// ---- 6. Depthwise conv K=9, VALID (padded tail written 0) ----
__global__ void k_dwconv(const float* __restrict__ h1, const float* __restrict__ dw,
                         float* __restrict__ h2) {
    size_t i = (size_t)blockIdx.x * blockDim.x + threadIdx.x;
    if (i >= (size_t)BSZ * HH * LL) return;
    int l = i & 4095, c = (int)((i >> 12) & 255), b = (int)(i >> 20);
    float acc = 0.f;
    if (l < LCC) {
        const float* in = h1 + ((size_t)b * HH + c) * LL + l;
#pragma unroll
        for (int k = 0; k < KK; ++k) acc += in[k] * dw[c * KK + k];
    }
    h2[((size_t)b * HH + c) * LL + l] = acc;
}

// ---- 7. BN stats per channel over (b, l<LC): atomic partial sums ----
__global__ void k_bnstats(const float* __restrict__ h2, float* __restrict__ st) {
    int c = blockIdx.x >> 4, part = blockIdx.x & 15;
    float s = 0.f, q = 0.f;
    const int total = BSZ * LCC;
    for (int i = part * 256 + threadIdx.x; i < total; i += 16 * 256) {
        int b = i / LCC, l = i % LCC;
        float v = h2[((size_t)b * HH + c) * LL + l];
        s += v; q += v * v;
    }
    __shared__ float s1[256], s2[256];
    s1[threadIdx.x] = s; s2[threadIdx.x] = q; __syncthreads();
    for (int k = 128; k > 0; k >>= 1) {
        if (threadIdx.x < k) { s1[threadIdx.x] += s1[threadIdx.x + k]; s2[threadIdx.x] += s2[threadIdx.x + k]; }
        __syncthreads();
    }
    if (threadIdx.x == 0) { atomicAdd(&st[c], s1[0]); atomicAdd(&st[HH + c], s2[0]); }
}

// ---- 8. BN + SiLU + repack f16 [b][kt][l][32] (tail zero-padded) ----
__global__ void k_bnsilu_pack(const float* __restrict__ h2, const float* __restrict__ st,
                              const float* __restrict__ g, const float* __restrict__ bb,
                              _Float16* __restrict__ Hp2) {
    size_t i = (size_t)blockIdx.x * blockDim.x + threadIdx.x;
    if (i >= (size_t)BSZ * HH * LL) return;
    int l = i & 4095, c = (int)((i >> 12) & 255), b = (int)(i >> 20);
    float out = 0.f;
    if (l < LCC) {
        const float inv = 1.f / (float)(BSZ * LCC);
        float m = st[c] * inv;
        float var = st[HH + c] * inv - m * m;
        float v = h2[((size_t)b * HH + c) * LL + l];
        v = (v - m) * rsqrtf(var + EPSF) * g[c] + bb[c];
        out = v / (1.f + __expf(-v));
    }
    Hp2[(((size_t)b * KT2 + (c >> 5)) * LL + l) * 32 + (c & 31)] = (_Float16)out;
}

// ---- 9. GEMM2 (WMMA f16) + bias ----
__global__ void k_gemm2(const _Float16* __restrict__ Ap, const _Float16* __restrict__ Hp2,
                        const float* __restrict__ b2, float* __restrict__ h3) {
    int lane = threadIdx.x & 31, wv = threadIdx.x >> 5;
    int bid = blockIdx.x;
    int nt = bid & 255, mtg = (bid >> 8) & 3, b = bid >> 10;
    int mt = mtg * 4 + wv;             // 0..15
    int half = lane >> 4, lr = lane & 15;
    int col = nt * 16 + lr;
    v8f acc = {};
    const _Float16* Bb = Hp2 + (((size_t)b * KT2) * LL + col) * 32;
    const _Float16* Aa = Ap + (((size_t)mt * KT2) * 16 + lr) * 32;
    for (int kt = 0; kt < KT2; ++kt) {
        if (kt + 1 < KT2) __builtin_prefetch(Bb + (size_t)(kt + 1) * LL * 32, 0, 1);
        v16h bf = load_frag_b(Bb + (size_t)kt * LL * 32, half);
        v16h af = load_frag_a(Aa + kt * 512, half);
        acc = wmma_f16(af, bf, acc);
    }
#pragma unroll
    for (int j = 0; j < 8; ++j) {
        int row = mt * 16 + j + 8 * half;
        h3[((size_t)b * HH + row) * LL + col] = acc[j] + b2[row];
    }
}

// ---- 10. Segment sum (atomics) ----
__global__ void k_segment(const float* __restrict__ h3, const int* __restrict__ wn,
                          float* __restrict__ sums, float* __restrict__ cnt) {
    int i = blockIdx.x;
    int b = i / LCC, lp = i % LCC;
    int idx = wn[b * LL + CSH + lp];     // 0..510
    int h = threadIdx.x;
    float v = h3[((size_t)b * HH + h) * LL + lp];
    atomicAdd(&sums[((size_t)b * NSEG + idx) * HH + h], v);
    if (h == 0) atomicAdd(&cnt[b * NSEG + idx], 1.f);
}

// ---- 11. node_out -> d_out[:,1:,:] (pre-z) + per-(b,h) sum ----
__global__ void k_nodeout(const float* __restrict__ sums, const float* __restrict__ cnt,
                          float* __restrict__ out, float* __restrict__ nodesum) {
    int b = blockIdx.x / NSEG, n = blockIdx.x % NSEG, h = threadIdx.x;
    float c = cnt[b * NSEG + n];
    float v = sums[((size_t)b * NSEG + n) * HH + h] / fmaxf(c, 1.f);
    out[((size_t)b * NN + 1 + n) * HH + h] = v;
    atomicAdd(&nodesum[b * HH + h], v);
}

// ---- 12. z1 = (vn + nodesum) @ w1^T ----
__global__ void k_z1(const float* __restrict__ xn, const float* __restrict__ nodesum,
                     const float* __restrict__ w1, float* __restrict__ z1) {
    int b = blockIdx.x, o = threadIdx.x;
    float acc = 0.f;
    for (int c = 0; c < HH; ++c)
        acc += (xn[((size_t)b * NN) * HH + c] + nodesum[b * HH + c]) * w1[(size_t)o * HH + c];
    z1[b * HH + o] = acc;
}

// ---- 13. BN2 stats over batch ----
__global__ void k_bn2stats(const float* __restrict__ z1, float* __restrict__ st) {
    int o = threadIdx.x;
    float s = 0.f, q = 0.f;
    for (int b = 0; b < BSZ; ++b) { float v = z1[b * HH + o]; s += v; q += v * v; }
    float m = s / BSZ;
    st[o] = m;
    st[HH + o] = q / BSZ - m * m;
}

// ---- 14. BN2 + ReLU ----
__global__ void k_bn2apply(const float* __restrict__ z1, const float* __restrict__ st,
                           const float* __restrict__ g, const float* __restrict__ bb,
                           float* __restrict__ zr) {
    int b = blockIdx.x, o = threadIdx.x;
    float v = (z1[b * HH + o] - st[o]) * rsqrtf(st[HH + o] + EPSF) * g[o] + bb[o];
    zr[b * HH + o] = fmaxf(v, 0.f);
}

// ---- 15. z2 = zr @ w2^T ; write d_out[:,0,:] ----
__global__ void k_z2(const float* __restrict__ zr, const float* __restrict__ w2,
                     float* __restrict__ out, float* __restrict__ z2) {
    int b = blockIdx.x, o = threadIdx.x;
    float acc = 0.f;
    for (int c = 0; c < HH; ++c) acc += zr[b * HH + c] * w2[(size_t)o * HH + c];
    z2[b * HH + o] = acc;
    out[(size_t)b * NN * HH + o] = acc;
}

// ---- 16. broadcast add z2 to node rows ----
__global__ void k_addz(const float* __restrict__ z2, float* __restrict__ out) {
    int b = blockIdx.x / NSEG, n = blockIdx.x % NSEG, h = threadIdx.x;
    out[((size_t)b * NN + 1 + n) * HH + h] += z2[b * HH + h];
}

extern "C" void kernel_launch(void* const* d_in, const int* in_sizes, int n_in,
                              void* d_out, int out_size, void* d_ws, size_t ws_size,
                              hipStream_t stream) {
    const float* x    = (const float*)d_in[0];
    const float* ef   = (const float*)d_in[1];
    const int*   wn   = (const int*)d_in[2];
    const int*   we   = (const int*)d_in[3];
    const float* enc  = (const float*)d_in[4];
    const float* ln_g = (const float*)d_in[5];
    const float* ln_b = (const float*)d_in[6];
    const float* pc1w = (const float*)d_in[7];
    const float* pc1b = (const float*)d_in[8];
    const float* dcw  = (const float*)d_in[9];
    const float* bng  = (const float*)d_in[10];
    const float* bnb  = (const float*)d_in[11];
    const float* pc2w = (const float*)d_in[12];
    const float* pc2b = (const float*)d_in[13];
    const float* w1   = (const float*)d_in[14];
    const float* bn2g = (const float*)d_in[15];
    const float* bn2b = (const float*)d_in[16];
    const float* w2   = (const float*)d_in[17];
    float* out = (float*)d_out;
    char* ws = (char*)d_ws;

    // workspace layout (256B aligned)
    const size_t o_xn   = 0;                                  // 16,777,216
    const size_t o_Hp1  = o_xn  + (size_t)BSZ*NN*HH*4;        // 109,051,904
    const size_t o_Ap1  = o_Hp1 + (size_t)BSZ*KT1*LL*32*2;    // 425,984
    const size_t o_Ap2  = o_Ap1 + (size_t)32*KT1*16*32*2;     // 131,072
    const size_t o_h1   = o_Ap2 + (size_t)16*KT2*16*32*2;     // 134,217,728 (reused as h3)
    const size_t o_h2   = o_h1  + (size_t)BSZ*HH*LL*4;        // 134,217,728
    const size_t o_Hp2  = o_h2  + (size_t)BSZ*HH*LL*4;        // 67,108,864
    const size_t o_sums = o_Hp2 + (size_t)BSZ*KT2*LL*32*2;    // 16,744,448
    const size_t o_cnt  = o_sums + (size_t)BSZ*NSEG*HH*4;     // 65,536 (rounded)
    const size_t o_bnst = o_cnt + 65536;                      // 2,048
    const size_t o_nsum = o_bnst + 2048;                      // 32,768
    const size_t o_z1   = o_nsum + 32768;
    const size_t o_zr   = o_z1 + 32768;
    const size_t o_st2  = o_zr + 32768;
    const size_t o_z2   = o_st2 + 2048;

    float*    xn   = (float*)(ws + o_xn);
    _Float16* Hp1  = (_Float16*)(ws + o_Hp1);
    _Float16* Ap1  = (_Float16*)(ws + o_Ap1);
    _Float16* Ap2  = (_Float16*)(ws + o_Ap2);
    float*    h1   = (float*)(ws + o_h1);
    float*    h2   = (float*)(ws + o_h2);
    _Float16* Hp2  = (_Float16*)(ws + o_Hp2);
    float*    h3   = (float*)(ws + o_h1);       // alias: h1 dead after conv
    float*    sums = (float*)(ws + o_sums);
    float*    cnt  = (float*)(ws + o_cnt);
    float*    bnst = (float*)(ws + o_bnst);
    float*    nsum = (float*)(ws + o_nsum);
    float*    z1   = (float*)(ws + o_z1);
    float*    zr   = (float*)(ws + o_zr);
    float*    st2  = (float*)(ws + o_st2);
    float*    z2   = (float*)(ws + o_z2);

    k_layernorm<<<BSZ * NN, HH, 0, stream>>>(x, ln_g, ln_b, xn);
    k_pack_w1<<<(32 * KT1 * 16 * 32 + 255) / 256, 256, 0, stream>>>(pc1w, Ap1);
    k_pack_w2<<<(16 * KT2 * 16 * 32 + 255) / 256, 256, 0, stream>>>(pc2w, Ap2);
    k_build_h<<<BSZ * LL, 128, 0, stream>>>(xn, ef, wn, we, enc, Hp1);
    k_gemm1_glu<<<BSZ * 4 * 256, 128, 0, stream>>>(Ap1, Hp1, pc1b, h1);
    k_dwconv<<<(int)(((size_t)BSZ * HH * LL) / 256), 256, 0, stream>>>(h1, dcw, h2);

    // zero sums..nodesum region (covers bnst too) before stats/segment
    hipMemsetAsync(ws + o_sums, 0, (o_nsum + 32768) - o_sums, stream);

    k_bnstats<<<HH * 16, 256, 0, stream>>>(h2, bnst);
    k_bnsilu_pack<<<(int)(((size_t)BSZ * HH * LL) / 256), 256, 0, stream>>>(h2, bnst, bng, bnb, Hp2);
    k_gemm2<<<BSZ * 4 * 256, 128, 0, stream>>>(Ap2, Hp2, pc2b, h3);
    k_segment<<<BSZ * LCC, HH, 0, stream>>>(h3, wn, sums, cnt);
    k_nodeout<<<BSZ * NSEG, HH, 0, stream>>>(sums, cnt, out, nsum);
    k_z1<<<BSZ, HH, 0, stream>>>(xn, nsum, w1, z1);
    k_bn2stats<<<1, HH, 0, stream>>>(z1, st2);
    k_bn2apply<<<BSZ, HH, 0, stream>>>(z1, st2, bn2g, bn2b, zr);
    k_z2<<<BSZ, HH, 0, stream>>>(zr, w2, out, z2);
    k_addz<<<BSZ * NSEG, HH, 0, stream>>>(z2, out);
}
